// RPN_38792144618253
// MI455X (gfx1250) — matrix operand order repaired and checked by reading, not compiled, not run on hardware
//
#include <hip/hip_runtime.h>

// ---------------------------------------------------------------------------
// Types for CDNA5 WMMA (wave32, v_wmma_f32_16x16x32_bf16)
// ---------------------------------------------------------------------------
typedef __attribute__((ext_vector_type(16))) __bf16   v16bf;
typedef __attribute__((ext_vector_type(8)))  float    v8f;
typedef __attribute__((ext_vector_type(4)))  unsigned u32x4;

union FragU { u32x4 u[2]; v16bf v; };

__device__ inline unsigned short f2bf(float f) {
  unsigned u = __builtin_bit_cast(unsigned, f);
  u += 0x7fffu + ((u >> 16) & 1u);         // round-to-nearest-even
  return (unsigned short)(u >> 16);
}

#define NEG_INF (-__builtin_inff())

// problem sizes (fixed by the reference)
#define BATCH 32
#define CIN   256
#define MID   256
#define HH    64
#define WW    64
#define NPOS  (BATCH * HH * WW)     // 131072 spatial positions
#define APOS  (HH * WW * 9)         // 36864 anchors per image
#define NANCH (BATCH * APOS)        // 1179648
#define NOUT  300
#define NMS_CAP 6144

// ---------------------------------------------------------------------------
// K0: x (f32 NCHW) -> bf16 NHWC  (channel-contiguous for implicit-GEMM A)
// ---------------------------------------------------------------------------
__global__ void k_prep_x(const float* __restrict__ X, unsigned short* __restrict__ XH) {
  const int TOT = BATCH * CIN * HH * WW;   // 33,554,432
  for (int idx = blockIdx.x * blockDim.x + threadIdx.x; idx < TOT;
       idx += gridDim.x * blockDim.x) {
    int c   = idx & 255;
    int pos = idx >> 8;
    int xx = pos & 63, yy = (pos >> 6) & 63, bb = pos >> 12;
    float v = X[((size_t)((bb << 8) + c) << 12) + (yy << 6) + xx];
    XH[idx] = f2bf(v);
  }
}

// ---------------------------------------------------------------------------
// K1: weight transforms + anchors + roi_indices (all tiny, one grid-stride kernel)
//   WB : conv3 weights as bf16 [K=2304][M=256], K = tap*256 + cin
//   W1 : 1x1 weights as bf16 [K=256][N=64] (N: 0..35 = loc, 36..53 = score, pad 0)
// ---------------------------------------------------------------------------
__global__ void k_prep_misc(const float* __restrict__ Wc, const float* __restrict__ Ws,
                            const float* __restrict__ Wl,
                            unsigned short* __restrict__ WB, unsigned short* __restrict__ W1,
                            float* __restrict__ anch_out, float* __restrict__ ridx_out) {
  const int TOT_WB = 2304 * 256;
  const int TOT_W1 = 256 * 64;
  const int TOT_AN = APOS * 4;
  const int TOT_RI = BATCH * NOUT;
  const int TOT = TOT_WB + TOT_W1 + TOT_AN + TOT_RI;
  for (int idx = blockIdx.x * blockDim.x + threadIdx.x; idx < TOT;
       idx += gridDim.x * blockDim.x) {
    if (idx < TOT_WB) {
      int k = idx >> 8, m = idx & 255;
      int t = k >> 8, c = k & 255;
      int ky = t / 3, kx = t - ky * 3;
      WB[idx] = f2bf(Wc[(size_t)m * 2304 + c * 9 + ky * 3 + kx]);
    } else if (idx < TOT_WB + TOT_W1) {
      int q = idx - TOT_WB;
      int c = q >> 6, nn = q & 63;
      float v = 0.0f;
      if (nn < 36)      v = Wl[(size_t)nn * 256 + c];
      else if (nn < 54) v = Ws[(size_t)(nn - 36) * 256 + c];
      W1[q] = f2bf(v);
    } else if (idx < TOT_WB + TOT_W1 + TOT_AN) {
      int q = idx - TOT_WB - TOT_W1;
      int j = q >> 2, comp = q & 3;
      int pos = j / 9, a = j - pos * 9;
      int yy = pos >> 6, xx = pos & 63;
      const float ratios[3] = {0.5f, 1.0f, 2.0f};
      const float scales[3] = {8.0f, 16.0f, 32.0f};
      float rr = ratios[a / 3], ss = scales[a % 3];
      float hh = 16.0f * ss * sqrtf(rr);
      float wwid = 16.0f * ss * sqrtf(1.0f / rr);
      float cx = xx * 32.0f + 8.0f, cy = yy * 32.0f + 8.0f;
      float o = (comp == 0) ? (cx - 0.5f * wwid)
              : (comp == 1) ? (cy - 0.5f * hh)
              : (comp == 2) ? (cx + 0.5f * wwid)
                            : (cy + 0.5f * hh);
      anch_out[q] = o;
    } else {
      int q = idx - TOT_WB - TOT_W1 - TOT_AN;
      ridx_out[q] = (float)(q / NOUT);
    }
  }
}

// ---------------------------------------------------------------------------
// K2: 3x3 conv as implicit-GEMM bf16 WMMA, fused bias + ReLU.
//   GEMM: M = spatial positions, N = 256 out-channels, K = 2304 (9 taps x 256).
//   Block = 256 threads (8 waves), covers 256 positions x 64 channels.
//   Wave  = 2 M-tiles x 4 N-tiles (8 accumulators, B reused 2x).
//   Weights staged per-block into double-buffered LDS (4 KB chunks) so the
//   8 waves share one global read; next chunk + next A fragments prefetched
//   into registers while the current 8 WMMAs issue.
// ---------------------------------------------------------------------------
__global__ __launch_bounds__(256) void k_conv3(const unsigned short* __restrict__ XH_,
                                               const unsigned short* __restrict__ WB_,
                                               const float* __restrict__ bc,
                                               unsigned short* __restrict__ H_) {
  __shared__ alignas(16) unsigned short wbuf[2][32 * 64];   // 2 x 4 KB
  const char* XH = (const char*)XH_;
  const char* WB = (const char*)WB_;
  char*       Hd = (char*)H_;
  const int tid  = threadIdx.x;
  const int wave = tid >> 5, lane = tid & 31;
  const int l15  = lane & 15;
  const int hisel = (lane >= 16) ? 16 : 0;
  const int n0    = blockIdx.y * 64;
  const int pbase = blockIdx.x * 256;

  // this lane's two M-tile positions (32 consecutive positions, one image row)
  const int p0   = pbase + (wave * 2) * 16 + l15;
  const int bimg = p0 >> 12;
  const int py   = (p0 >> 6) & 63;
  int px[2];
  px[0] = p0 & 63;
  px[1] = (p0 + 16) & 63;

  const u32x4 z4 = {0u, 0u, 0u, 0u};
  const v8f   z8 = {0, 0, 0, 0, 0, 0, 0, 0};
  v8f acc[2][4] = {{z8, z8, z8, z8}, {z8, z8, z8, z8}};

  const int krow = tid >> 3, cg = tid & 7;   // cooperative weight staging coords

  // A-fragment loader for k-step kk (t = kk/8 tap, cc = kk%8 channel chunk)
  auto loadA = [&](int kk, FragU* fa2) {
    const int t  = kk >> 3, cc = kk & 7;
    const int ky = t / 3, kx = t - ky * 3;
    const int yy = py + ky - 1;
#pragma unroll
    for (int m = 0; m < 2; ++m) {
      const int xx = px[m] + kx - 1;
      fa2[m].u[0] = z4; fa2[m].u[1] = z4;
      if (!(((unsigned)yy > 63u) || ((unsigned)xx > 63u))) {   // SAME padding
        const char* ap = XH + (size_t)((bimg << 12) + (yy << 6) + xx) * 512
                            + cc * 64 + hisel;
        fa2[m].u[0] = *(const u32x4*)ap;         // b128
        fa2[m].u[1] = *(const u32x4*)(ap + 32);  // b128
      }
    }
  };

  // stage weight chunk 0 into LDS, prefetch A for k=0
  {
    const u32x4 w0 = *(const u32x4*)(WB + (size_t)krow * 512 + (size_t)n0 * 2 + cg * 16);
    *(u32x4*)((char*)wbuf[0] + krow * 128 + cg * 16) = w0;
  }
  FragU fa[2][2];
  loadA(0, fa[0]);
  __syncthreads();

#pragma unroll 2
  for (int k = 0; k < 72; ++k) {
    const int cur = k & 1, nxt = cur ^ 1;
    const bool more = (k + 1) < 72;
    // prefetch next weight chunk (to regs) and next A fragments
    u32x4 wstage = z4;
    if (more) {
      wstage = *(const u32x4*)(WB + (size_t)((k + 1) * 32 + krow) * 512
                                  + (size_t)n0 * 2 + cg * 16);
      loadA(k + 1, fa[nxt]);
    }
    // B fragments from LDS, 8 WMMAs (2 M x 4 N)
    const char* wb = (const char*)wbuf[cur] + (hisel >> 1) * 128 + l15 * 2;
#pragma unroll
    for (int nt = 0; nt < 4; ++nt) {
      FragU fb;
#pragma unroll
      for (int s = 0; s < 16; ++s)
        fb.v[s] = *(const __bf16*)(wb + (s < 8 ? s : 8 + s) * 128 + nt * 32);
#pragma unroll
      for (int m = 0; m < 2; ++m)
        acc[m][nt] = __builtin_amdgcn_wmma_f32_16x16x32_bf16(
            false, fa[cur][m].v, false, fb.v, (short)0, acc[m][nt], false, false);
    }
    __syncthreads();
    if (more)
      *(u32x4*)((char*)wbuf[nxt] + krow * 128 + cg * 16) = wstage;
    __syncthreads();
  }

  // fused bias + ReLU, store h as bf16 NHWC (A-layout for the 1x1 GEMM)
#pragma unroll
  for (int m = 0; m < 2; ++m) {
    const int ptbase = pbase + (wave * 2 + m) * 16;
#pragma unroll
    for (int nt = 0; nt < 4; ++nt) {
      const int   n  = n0 + nt * 16 + l15;
      const float bb = bc[n];
#pragma unroll
      for (int v = 0; v < 8; ++v) {
        const int pos = ptbase + v + (hisel ? 8 : 0);
        float val = fmaxf(acc[m][nt][v] + bb, 0.0f);
        *(unsigned short*)(Hd + (size_t)pos * 512 + n * 2) = f2bf(val);
      }
    }
  }
}

// ---------------------------------------------------------------------------
// K3: fused 1x1 convs (loc 36ch + score 18ch in one N=64 GEMM), bias added,
//     f32 results written straight into d_out in the reference's flat layout.
//     Whole W1 (32 KB) staged once into LDS; B fragments read one bf16 element at a time.
// ---------------------------------------------------------------------------
__global__ __launch_bounds__(256) void k_conv1(const unsigned short* __restrict__ H_,
                                               const unsigned short* __restrict__ W1_,
                                               const float* __restrict__ bl,
                                               const float* __restrict__ bs,
                                               float* __restrict__ locs_out,
                                               float* __restrict__ scores_out) {
  __shared__ alignas(16) unsigned short w1s[256 * 64];   // 32 KB
  const char* Hd = (const char*)H_;
  const int tid  = threadIdx.x;
  const int wave = tid >> 5, lane = tid & 31;
  const int l15  = lane & 15;
  const int hisel = (lane >= 16) ? 16 : 0;
  const int ptile = blockIdx.x * 8 + wave;
  const int p     = ptile * 16 + l15;

  // cooperative stage of W1: 2048 b128 chunks / 256 threads = 8 each
#pragma unroll
  for (int i = 0; i < 8; ++i) {
    const int e = (tid + i * 256) * 16;   // byte offset
    *(u32x4*)((char*)w1s + e) = *(const u32x4*)((const char*)W1_ + e);
  }
  __syncthreads();

  const v8f z8 = {0, 0, 0, 0, 0, 0, 0, 0};
  v8f acc[4] = {z8, z8, z8, z8};

  for (int kk = 0; kk < 8; ++kk) {
    FragU fa;
    const char* ap = Hd + (size_t)p * 512 + kk * 64 + hisel;
    fa.u[0] = *(const u32x4*)ap;
    fa.u[1] = *(const u32x4*)(ap + 32);
    const char* bp = (const char*)w1s + (size_t)(kk * 32 + (hisel >> 1)) * 128 + l15 * 2;
#pragma unroll
    for (int nt = 0; nt < 4; ++nt) {
      FragU fb;
#pragma unroll
      for (int s = 0; s < 16; ++s)
        fb.v[s] = *(const __bf16*)(bp + (s < 8 ? s : 8 + s) * 128 + nt * 32);
      acc[nt] = __builtin_amdgcn_wmma_f32_16x16x32_bf16(
          false, fa.v, false, fb.v, (short)0, acc[nt], false, false);
    }
  }
#pragma unroll
  for (int nt = 0; nt < 4; ++nt) {
    const int n = nt * 16 + l15;
    float bias = 0.0f;
    if (n < 36)      bias = bl[n];
    else if (n < 54) bias = bs[n - 36];
#pragma unroll
    for (int v = 0; v < 8; ++v) {
      const int pos = ptile * 16 + v + (hisel ? 8 : 0);
      const float val = acc[nt][v] + bias;
      if (n < 36)      locs_out[(size_t)pos * 36 + n]          = val;  // (B,HWA,4) flat
      else if (n < 54) scores_out[(size_t)pos * 18 + (n - 36)] = val;  // (B,HWA,2) flat
    }
  }
}

// ---------------------------------------------------------------------------
// K4: softmax(fg) + loc2bbox decode + clip + min-size, per-anchor.
// ---------------------------------------------------------------------------
__global__ void k_decode(const float* __restrict__ locs, const float* __restrict__ scores,
                         const float* __restrict__ anchors,
                         const int* __restrict__ pih, const int* __restrict__ piw,
                         float* __restrict__ BX, float* __restrict__ SC) {
  const float img_h = (float)pih[0];
  const float img_w = (float)piw[0];
  for (int i = blockIdx.x * blockDim.x + threadIdx.x; i < NANCH;
       i += gridDim.x * blockDim.x) {
    const int j = i % APOS;                      // anchor index within image
    const float dx = locs[(size_t)i * 4 + 0];
    const float dy = locs[(size_t)i * 4 + 1];
    const float dw = locs[(size_t)i * 4 + 2];
    const float dh = locs[(size_t)i * 4 + 3];
    const float a0 = anchors[(size_t)j * 4 + 0];
    const float a1 = anchors[(size_t)j * 4 + 1];
    const float a2 = anchors[(size_t)j * 4 + 2];
    const float a3 = anchors[(size_t)j * 4 + 3];
    const float aw = a2 - a0, ah = a3 - a1;
    const float acx = a0 + 0.5f * aw, acy = a1 + 0.5f * ah;
    const float cx = dx * aw + acx, cy = dy * ah + acy;
    const float w = expf(dw) * aw, h = expf(dh) * ah;
    float x1 = cx - 0.5f * w, y1 = cy - 0.5f * h;
    float x2 = cx + 0.5f * w, y2 = cy + 0.5f * h;
    x1 = fminf(fmaxf(x1, 0.0f), img_w);  x2 = fminf(fmaxf(x2, 0.0f), img_w);
    y1 = fminf(fmaxf(y1, 0.0f), img_h);  y2 = fminf(fmaxf(y2, 0.0f), img_h);
    const bool ok = ((x2 - x1) >= 16.0f) && ((y2 - y1) >= 16.0f);
    const int pos = i / 9, a = i % 9;
    const float s0 = scores[(size_t)pos * 18 + 2 * a];
    const float s1 = scores[(size_t)pos * 18 + 2 * a + 1];
    const float fg = 1.0f / (1.0f + expf(s0 - s1));   // softmax class-1
    SC[i] = ok ? fg : NEG_INF;
    ((float4*)BX)[i] = make_float4(x1, y1, x2, y2);
  }
}

// ---------------------------------------------------------------------------
// K5: per-image top-6000 (binary-search threshold) + 300-round greedy NMS,
//     candidates compacted into LDS (~123 KB of the 320 KB WGP LDS).
// ---------------------------------------------------------------------------
__global__ __launch_bounds__(256) void k_nms(const float* __restrict__ SC,
                                             const float* __restrict__ BX,
                                             float* __restrict__ rois,
                                             float* __restrict__ mask) {
  extern __shared__ float smem[];
  float4* lbox = (float4*)smem;                  // NMS_CAP * 16 B
  float*  lsc  = smem + NMS_CAP * 4;             // NMS_CAP * 4 B
  __shared__ float redv[256];
  __shared__ int   redi[256];
  __shared__ float bc5[5];
  __shared__ int   s_cnt;

  const int b   = blockIdx.x;
  const int tid = threadIdx.x;
  const float*  sg = SC + (size_t)b * APOS;
  const float4* bg = (const float4*)BX + (size_t)b * APOS;

  // 6000th-score threshold via binary search on [0,1]
  float lo = 0.0f, hi = 1.0f;
  for (int it = 0; it < 24; ++it) {
    const float mid = 0.5f * (lo + hi);
    if (tid == 0) s_cnt = 0;
    __syncthreads();
    int c = 0;
    for (int i = tid; i < APOS; i += 256) if (sg[i] >= mid) ++c;
    atomicAdd(&s_cnt, c);
    __syncthreads();
    const int total = s_cnt;
    __syncthreads();
    if (total >= 6000) lo = mid; else hi = mid;
  }

  // compact survivors into LDS
  if (tid == 0) s_cnt = 0;
  __syncthreads();
  for (int i = tid; i < APOS; i += 256) {
    const float s = sg[i];
    if (s >= lo) {
      const int j = atomicAdd(&s_cnt, 1);
      if (j < NMS_CAP) { lsc[j] = s; lbox[j] = bg[i]; }
    }
  }
  __syncthreads();
  const int n = (s_cnt < NMS_CAP) ? s_cnt : NMS_CAP;

  for (int r = 0; r < NOUT; ++r) {
    // argmax over remaining scores
    float bv = NEG_INF; int bi = -1;
    for (int j = tid; j < n; j += 256)
      if (lsc[j] > bv) { bv = lsc[j]; bi = j; }
    redv[tid] = bv; redi[tid] = bi;
    __syncthreads();
    for (int s = 128; s > 0; s >>= 1) {
      if (tid < s && redv[tid + s] > redv[tid]) {
        redv[tid] = redv[tid + s]; redi[tid] = redi[tid + s];
      }
      __syncthreads();
    }
    if (tid == 0) {
      const float best = redv[0]; const int idx = redi[0];
      const int has = (best > NEG_INF) && (idx >= 0);
      const float4 bb = has ? lbox[idx] : make_float4(0, 0, 0, 0);
      const size_t ro = ((size_t)b * NOUT + r) * 4;
      rois[ro + 0] = bb.x; rois[ro + 1] = bb.y;
      rois[ro + 2] = bb.z; rois[ro + 3] = bb.w;
      mask[(size_t)b * NOUT + r] = has ? 1.0f : 0.0f;
      bc5[0] = bb.x; bc5[1] = bb.y; bc5[2] = bb.z; bc5[3] = bb.w;
      bc5[4] = (float)has;
    }
    __syncthreads();
    if (bc5[4] != 0.0f) {
      const float ax1 = bc5[0], ay1 = bc5[1], ax2 = bc5[2], ay2 = bc5[3];
      const float aarea = fmaxf(ax2 - ax1, 0.0f) * fmaxf(ay2 - ay1, 0.0f);
      for (int j = tid; j < n; j += 256) {
        if (lsc[j] > NEG_INF) {
          const float4 q = lbox[j];
          const float qarea = fmaxf(q.z - q.x, 0.0f) * fmaxf(q.w - q.y, 0.0f);
          const float ix1 = fmaxf(ax1, q.x), iy1 = fmaxf(ay1, q.y);
          const float ix2 = fminf(ax2, q.z), iy2 = fminf(ay2, q.w);
          const float inter = fmaxf(ix2 - ix1, 0.0f) * fmaxf(iy2 - iy1, 0.0f);
          const float iou = inter / (qarea + aarea - inter + 1e-9f);
          if (iou > 0.7f) lsc[j] = NEG_INF;   // suppress (also kills the seed)
        }
      }
    }
    __syncthreads();
  }
}

// ---------------------------------------------------------------------------
// launcher
// ---------------------------------------------------------------------------
extern "C" void kernel_launch(void* const* d_in, const int* in_sizes, int n_in,
                              void* d_out, int out_size, void* d_ws, size_t ws_size,
                              hipStream_t stream) {
  (void)in_sizes; (void)n_in; (void)out_size; (void)ws_size;
  const float* X   = (const float*)d_in[0];
  const int*   pih = (const int*)d_in[1];
  const int*   piw = (const int*)d_in[2];
  const float* Wc  = (const float*)d_in[3];
  const float* bcv = (const float*)d_in[4];
  const float* Ws  = (const float*)d_in[5];
  const float* bs  = (const float*)d_in[6];
  const float* Wl  = (const float*)d_in[7];
  const float* bl  = (const float*)d_in[8];

  float* outF = (float*)d_out;   // tuple outputs concatenated flat
  float* o_scores = outF + 0;          // 2,359,296
  float* o_locs   = outF + 2359296;    // 4,718,592
  float* o_rois   = outF + 7077888;    //    38,400
  float* o_ridx   = outF + 7116288;    //     9,600
  float* o_anch   = outF + 7125888;    //   147,456
  float* o_mask   = outF + 7273344;    //     9,600

  char* ws = (char*)d_ws;
  unsigned short* XH = (unsigned short*)(ws);              // 67,108,864 B (bf16 NHWC x)
  unsigned short* WB = (unsigned short*)(ws + 67108864);   //  1,179,648 B
  unsigned short* W1 = (unsigned short*)(ws + 68288512);   //     32,768 B
  unsigned short* Hm = (unsigned short*)(ws + 68321280);   // 67,108,864 B (bf16 NHWC h)
  float* SC = (float*)(ws);                                // reuses XH region (dead post-conv3)
  float* BX = (float*)(ws + 4718592);

  k_prep_x   <<<16384, 256, 0, stream>>>(X, XH);
  k_prep_misc<<<1024,  256, 0, stream>>>(Wc, Ws, Wl, WB, W1, o_anch, o_ridx);
  k_conv3    <<<dim3(512, 4, 1), 256, 0, stream>>>(XH, WB, bcv, Hm);
  k_conv1    <<<1024,  256, 0, stream>>>(Hm, W1, bl, bs, o_locs, o_scores);
  k_decode   <<<2048,  256, 0, stream>>>(o_locs, o_scores, o_anch, pih, piw, BX, SC);
  const size_t smem = (size_t)NMS_CAP * 16 + (size_t)NMS_CAP * 4;   // 122,880 B
  k_nms      <<<32,    256, smem, stream>>>(SC, BX, o_rois, o_mask);
}
